// Self_Attention_44160853737538
// MI455X (gfx1250) — matrix operand order repaired and checked by reading, not compiled, not run on hardware
//
#include <hip/hip_runtime.h>
#include <stdint.h>

#define EMB   1024
#define NHEAD 16
#define HEAD  64
#define BATCH 4
#define SEQ   2048
#define MROWS (BATCH*SEQ)   // 8192
#define WLPAD 72            // GEMM LDS row pad (elements): 144B stride, 16B aligned, conflict-free
#define PPAD  40            // attention P-tile row pad: 80B stride

typedef __bf16 bf16_t;
typedef bf16_t v16bf __attribute__((ext_vector_type(16)));
typedef float  v8f   __attribute__((ext_vector_type(8)));
typedef int    v4i_  __attribute__((ext_vector_type(4)));

#if __has_builtin(__builtin_amdgcn_global_load_async_to_lds_b128)
#define ASYNC_OK 1
#else
#define ASYNC_OK 0
#endif

union Frag {
    v16bf v;
    uint4 q[2];
};

__device__ __forceinline__ unsigned short f2bf(float x) {
    union { float f; unsigned u; } c; c.f = x;
    unsigned r = c.u + 0x7FFFu + ((c.u >> 16) & 1u);   // RNE
    return (unsigned short)(r >> 16);
}

// A-fragment (16x32 bf16): lane L%16 = row M; K = g*16 + (L/16)*8 + e
__device__ __forceinline__ v16bf load_frag_a(const unsigned short* rowp, int k0, int half) {
    Frag f;
    f.q[0] = *(const uint4*)(rowp + k0 + half * 8);
    f.q[1] = *(const uint4*)(rowp + k0 + 16 + half * 8);
    return f.v;
}

// B-fragment (32x16 bf16) from contiguous-K source: lane L%16 = col N; K = g*16 + (L/16)*8 + e
__device__ __forceinline__ v16bf load_frag_b(const unsigned short* colp, int k0, int half) {
    Frag f;
    f.q[0] = *(const uint4*)(colp + k0 + half * 8);
    f.q[1] = *(const uint4*)(colp + k0 + 16 + half * 8);
    return f.v;
}

// B-fragment from LDS tile (base already includes row*WLPAD + kk)
__device__ __forceinline__ v16bf load_frag_b_lds(const unsigned short* colp, int half) {
    Frag f;
    f.q[0] = *(const uint4*)(colp + half * 8);
    f.q[1] = *(const uint4*)(colp + 16 + half * 8);
    return f.v;
}

__device__ __forceinline__ v8f wmma_bf16(v16bf a, v16bf b, v8f c) {
    return __builtin_amdgcn_wmma_f32_16x16x32_bf16(false, a, false, b, (short)0, c, false, false);
}

// async (or fallback sync) 16B global -> LDS copy
__device__ __forceinline__ void copy16_g2l(const unsigned short* src, unsigned short* dst) {
#if ASYNC_OK
    __builtin_amdgcn_global_load_async_to_lds_b128((v4i_*)src, (v4i_*)dst, 0, 0);
#else
    *(uint4*)dst = *(const uint4*)src;
#endif
}
__device__ __forceinline__ void wait_async0() {
#if ASYNC_OK
    asm volatile("s_wait_asynccnt 0x0" ::: "memory");
#else
    asm volatile("s_wait_dscnt 0x0" ::: "memory");
#endif
}
// split barrier without the conservative loadcnt drain of __syncthreads()
__device__ __forceinline__ void block_sync() {
    asm volatile("s_barrier_signal -1\n\ts_barrier_wait -1" ::: "memory");
}

// cooperative: stage 64 rows x 64 k of Bt (contiguous-K weights) into padded LDS tile
__device__ __forceinline__ void issue_b_tile(const unsigned short* __restrict__ btBase,
                                             int k0, unsigned short* lbuf, int tid) {
    int row = tid >> 2;        // 0..63
    int c0  = tid & 3;
#pragma unroll
    for (int it = 0; it < 2; ++it) {
        int c = c0 + it * 4;   // 8 chunks x 8 elements = 64 k
        copy16_g2l(btBase + (size_t)row * EMB + k0 + c * 8, lbuf + row * WLPAD + c * 8);
    }
}

// one K-stage (64 deep): 16 WMMAs per wave
__device__ __forceinline__ void gemm_stage(const unsigned short* cur, int ln, int half,
                                           const v16bf a[2][2], v8f acc[2][4]) {
#pragma unroll
    for (int kk = 0; kk < 2; ++kk) {
#pragma unroll
        for (int j = 0; j < 4; ++j) {
            v16bf bfr = load_frag_b_lds(cur + (j * 16 + ln) * WLPAD + kk * 32, half);
            acc[0][j] = wmma_bf16(a[0][kk], bfr, acc[0][j]);
            acc[1][j] = wmma_bf16(a[1][kk], bfr, acc[1][j]);
        }
    }
}

__device__ __forceinline__ void load_a_stage(const unsigned short* arow0,
                                             const unsigned short* arow1,
                                             int k0, int half, v16bf a[2][2]) {
    a[0][0] = load_frag_a(arow0, k0, half);
    a[0][1] = load_frag_a(arow0, k0 + 32, half);
    a[1][0] = load_frag_a(arow1, k0, half);
    a[1][1] = load_frag_a(arow1, k0 + 32, half);
}

// ---------------- prep ----------------
__global__ void cvt_bf16_kernel(const float* __restrict__ src,
                                unsigned short* __restrict__ dst, int n8) {
    int i = blockIdx.x * blockDim.x + threadIdx.x;
    int stride = gridDim.x * blockDim.x;
    for (; i < n8; i += stride) {
        float4 f0 = ((const float4*)src)[i * 2];
        float4 f1 = ((const float4*)src)[i * 2 + 1];
        union { unsigned short u[8]; uint4 q; } o;
        o.u[0] = f2bf(f0.x); o.u[1] = f2bf(f0.y); o.u[2] = f2bf(f0.z); o.u[3] = f2bf(f0.w);
        o.u[4] = f2bf(f1.x); o.u[5] = f2bf(f1.y); o.u[6] = f2bf(f1.z); o.u[7] = f2bf(f1.w);
        ((uint4*)dst)[i] = o.q;
    }
}

// W[k][n] fp32 -> Wt[n][k] bf16, LDS-tiled 32x32 (both sides coalesced)
__global__ __launch_bounds__(256) void transpose_bf16_kernel(const float* __restrict__ W,
                                                             unsigned short* __restrict__ Wt) {
    __shared__ unsigned short tile[32][33];
    int bx = blockIdx.x & 31, by = blockIdx.x >> 5;
    int tx = threadIdx.x & 31, ty = threadIdx.x >> 5;   // 32 x 8
#pragma unroll
    for (int i = 0; i < 4; ++i)
        tile[ty + i * 8][tx] = f2bf(W[(size_t)(by * 32 + ty + i * 8) * EMB + bx * 32 + tx]);
    __syncthreads();
#pragma unroll
    for (int i = 0; i < 4; ++i)
        Wt[(size_t)(bx * 32 + ty + i * 8) * EMB + by * 32 + tx] = tile[tx][ty + i * 8];
}

// pipelined 256x64 GEMM body shared by qkv/proj: branch-free ping-pong, 16 stages of K=64
#define GEMM_PIPELINE(btBase, arow0, arow1, buf0, buf1, acc)                        \
    v16bf aE[2][2], aO[2][2];                                                       \
    issue_b_tile(btBase, 0, buf0, tid);                                             \
    load_a_stage(arow0, arow1, 0, half, aE);                                        \
    wait_async0(); block_sync();                                                    \
    for (int i = 0; i < 14; i += 2) {                                               \
        issue_b_tile(btBase, (i + 1) * 64, buf1, tid);                              \
        load_a_stage(arow0, arow1, (i + 1) * 64, half, aO);                         \
        gemm_stage(buf0, ln, half, aE, acc);                                        \
        wait_async0(); block_sync();                                                \
        issue_b_tile(btBase, (i + 2) * 64, buf0, tid);                              \
        load_a_stage(arow0, arow1, (i + 2) * 64, half, aE);                         \
        gemm_stage(buf1, ln, half, aO, acc);                                        \
        wait_async0(); block_sync();                                                \
    }                                                                               \
    issue_b_tile(btBase, 15 * 64, buf1, tid);                                       \
    load_a_stage(arow0, arow1, 15 * 64, half, aO);                                  \
    gemm_stage(buf0, ln, half, aE, acc);                                            \
    wait_async0(); block_sync();                                                    \
    gemm_stage(buf1, ln, half, aO, acc);

// ---------------- QKV projections ----------------
// block: 256 rows x 64 cols of one of Q/K/V; wave: 32x64 (8 accumulators)
__global__ __launch_bounds__(256) void qkv_kernel(
    const unsigned short* __restrict__ xb,
    const unsigned short* __restrict__ wtq, const unsigned short* __restrict__ wtk,
    const unsigned short* __restrict__ wtv,
    const float* __restrict__ bq, const float* __restrict__ bk, const float* __restrict__ bv,
    unsigned short* __restrict__ qo, unsigned short* __restrict__ ko,
    unsigned short* __restrict__ vto)
{
    __shared__ __align__(16) unsigned short lbuf[2][64][WLPAD];
    const int tid  = threadIdx.x;
    const int lane = tid & 31;
    const int half = lane >> 4;
    const int ln   = lane & 15;
    const int wv   = tid >> 5;

    int mt    = blockIdx.x / 48;
    int rest  = blockIdx.x % 48;
    int which = rest >> 4;     // 0=Q 1=K 2=V
    int h     = rest & 15;
    int wrow0 = mt * 256 + wv * 32;

    const unsigned short* wt   = (which == 0) ? wtq : (which == 1) ? wtk : wtv;
    const float*          bias = (which == 0) ? bq  : (which == 1) ? bk  : bv;
    const unsigned short* btBase = wt + (size_t)(h * 64) * EMB;
    const unsigned short* arow0 = xb + (size_t)(wrow0 + ln) * EMB;
    const unsigned short* arow1 = xb + (size_t)(wrow0 + 16 + ln) * EMB;

    v8f acc[2][4] = {};
    GEMM_PIPELINE(btBase, arow0, arow1, &lbuf[0][0][0], &lbuf[1][0][0], acc)

    int b  = wrow0 >> 11;
    int t0 = wrow0 & (SEQ - 1);
    size_t bh = (size_t)(b * NHEAD + h);
#pragma unroll
    for (int m = 0; m < 2; ++m) {
#pragma unroll
        for (int j = 0; j < 4; ++j) {
            float bval = bias[h * 64 + j * 16 + ln];
            int d = j * 16 + ln;
#pragma unroll
            for (int r = 0; r < 8; ++r) {
                int t = t0 + m * 16 + r + 8 * half;
                unsigned short o = f2bf(acc[m][j][r] + bval);
                if (which == 0)      qo [(bh * SEQ + t) * HEAD + d] = o;
                else if (which == 1) ko [(bh * SEQ + t) * HEAD + d] = o;
                else                 vto[(bh * HEAD + d) * SEQ + t] = o;
            }
        }
    }
}

// ---------------- flash attention ----------------
__global__ __launch_bounds__(256) void attn_kernel(
    const unsigned short* __restrict__ qb,
    const unsigned short* __restrict__ kbm,
    const unsigned short* __restrict__ vtb,
    unsigned short* __restrict__ attnb)
{
    __shared__ __align__(16) unsigned short lds_p[8][16][PPAD];
    const int lane = threadIdx.x & 31;
    const int half = lane >> 4;
    const int ln   = lane & 15;
    const int wv   = threadIdx.x >> 5;
    int wid = blockIdx.x * 8 + wv;
    int bh  = wid >> 7;
    int q0  = (wid & 127) * 16;

    const unsigned short* qrow = qb + ((size_t)bh * SEQ + q0 + ln) * HEAD;
    v16bf qa0 = load_frag_a(qrow, 0, half);
    v16bf qa1 = load_frag_a(qrow, 32, half);

    v8f o[4] = {};
    float mrow[8], lrow[8];
#pragma unroll
    for (int r = 0; r < 8; ++r) { mrow[r] = -1e30f; lrow[r] = 0.0f; }

    int nkb = (q0 + 47) >> 5;
    for (int kb2 = 0; kb2 < nkb; ++kb2) {
        int key0 = kb2 * 32;
        v8f s[2] = {};
#pragma unroll
        for (int tile = 0; tile < 2; ++tile) {
            const unsigned short* krow =
                kbm + ((size_t)bh * SEQ + key0 + tile * 16 + ln) * HEAD;
            s[tile] = wmma_bf16(qa0, load_frag_b(krow, 0, half), s[tile]);
            s[tile] = wmma_bf16(qa1, load_frag_b(krow, 32, half), s[tile]);
        }
        // prefetch V fragments now -> latency hides under softmax VALU
        v16bf vf[4];
#pragma unroll
        for (int j = 0; j < 4; ++j) {
            const unsigned short* vrow = vtb + ((size_t)bh * HEAD + j * 16 + ln) * SEQ;
            vf[j] = load_frag_b(vrow, key0, half);
        }
        // scale, causal mask, online softmax (row = q0 + r + 8*half, col = key0+tile*16+ln)
#pragma unroll
        for (int r = 0; r < 8; ++r) {
            int row = q0 + r + 8 * half;
            float sv0 = s[0][r] * 0.125f;
            float sv1 = s[1][r] * 0.125f;
            if (key0 + ln > row)      sv0 = -1e30f;
            if (key0 + 16 + ln > row) sv1 = -1e30f;
            float mx = fmaxf(sv0, sv1);
            mx = fmaxf(mx, __shfl_xor(mx, 1, 32));
            mx = fmaxf(mx, __shfl_xor(mx, 2, 32));
            mx = fmaxf(mx, __shfl_xor(mx, 4, 32));
            mx = fmaxf(mx, __shfl_xor(mx, 8, 32));
            float mnew  = fmaxf(mrow[r], mx);
            float alpha = __expf(mrow[r] - mnew);
            float e0 = __expf(sv0 - mnew);
            float e1 = __expf(sv1 - mnew);
            float sum = e0 + e1;
            sum += __shfl_xor(sum, 1, 32);
            sum += __shfl_xor(sum, 2, 32);
            sum += __shfl_xor(sum, 4, 32);
            sum += __shfl_xor(sum, 8, 32);
            lrow[r] = lrow[r] * alpha + sum;
            mrow[r] = mnew;
#pragma unroll
            for (int j = 0; j < 4; ++j) o[j][r] *= alpha;
            lds_p[wv][r + 8 * half][ln]      = f2bf(e0);
            lds_p[wv][r + 8 * half][16 + ln] = f2bf(e1);
        }
        asm volatile("s_wait_dscnt 0x0" ::: "memory");   // same-wave DS write -> DS read
        Frag pf;   // P in A layout: row = ln, K = 32 keys of this block
        const unsigned short* prow = &lds_p[wv][ln][0];
        pf.q[0] = *(const uint4*)(prow + half * 8);
        pf.q[1] = *(const uint4*)(prow + 16 + half * 8);
#pragma unroll
        for (int j = 0; j < 4; ++j)
            o[j] = wmma_bf16(pf.v, vf[j], o[j]);
    }
    int b = bh >> 4, h = bh & 15;
#pragma unroll
    for (int j = 0; j < 4; ++j) {
#pragma unroll
        for (int r = 0; r < 8; ++r) {
            int t = q0 + r + 8 * half;
            attnb[((size_t)(b * SEQ + t)) * EMB + h * HEAD + j * 16 + ln] =
                f2bf(o[j][r] / lrow[r]);
        }
    }
}

// ---------------- output projection (fp32 out) ----------------
__global__ __launch_bounds__(256) void proj_kernel(
    const unsigned short* __restrict__ ab,
    const unsigned short* __restrict__ wto,
    const float* __restrict__ bo,
    float* __restrict__ out)
{
    __shared__ __align__(16) unsigned short lbuf[2][64][WLPAD];
    const int tid  = threadIdx.x;
    const int lane = tid & 31;
    const int half = lane >> 4;
    const int ln   = lane & 15;
    const int wv   = tid >> 5;

    int mt    = blockIdx.x >> 4;
    int nt    = blockIdx.x & 15;
    int wrow0 = mt * 256 + wv * 32;
    const unsigned short* btBase = wto + (size_t)(nt * 64) * EMB;
    const unsigned short* arow0 = ab + (size_t)(wrow0 + ln) * EMB;
    const unsigned short* arow1 = ab + (size_t)(wrow0 + 16 + ln) * EMB;

    v8f acc[2][4] = {};
    GEMM_PIPELINE(btBase, arow0, arow1, &lbuf[0][0][0], &lbuf[1][0][0], acc)

#pragma unroll
    for (int m = 0; m < 2; ++m) {
#pragma unroll
        for (int j = 0; j < 4; ++j) {
            int n = nt * 64 + j * 16 + ln;
            float bval = bo[n];
#pragma unroll
            for (int r = 0; r < 8; ++r)
                out[(size_t)(wrow0 + m * 16 + r + 8 * half) * EMB + n] = acc[m][j][r] + bval;
        }
    }
}

extern "C" void kernel_launch(void* const* d_in, const int* in_sizes, int n_in,
                              void* d_out, int out_size, void* d_ws, size_t ws_size,
                              hipStream_t stream) {
    const float* x  = (const float*)d_in[0];
    const float* Wq = (const float*)d_in[1];
    const float* bq = (const float*)d_in[2];
    const float* Wk = (const float*)d_in[3];
    const float* bk = (const float*)d_in[4];
    const float* Wv = (const float*)d_in[5];
    const float* bv = (const float*)d_in[6];
    const float* Wo = (const float*)d_in[7];
    const float* bo = (const float*)d_in[8];
    float* out = (float*)d_out;

    unsigned char* ws = (unsigned char*)d_ws;
    size_t off = 0;
    auto alloc = [&](size_t bytes) -> void* {
        void* p = ws + off;
        off += (bytes + 255) & ~(size_t)255;
        return p;
    };
    unsigned short* xb  = (unsigned short*)alloc((size_t)MROWS * EMB * 2);
    unsigned short* wtq = (unsigned short*)alloc((size_t)EMB * EMB * 2);
    unsigned short* wtk = (unsigned short*)alloc((size_t)EMB * EMB * 2);
    unsigned short* wtv = (unsigned short*)alloc((size_t)EMB * EMB * 2);
    unsigned short* wto = (unsigned short*)alloc((size_t)EMB * EMB * 2);
    unsigned short* qb  = (unsigned short*)alloc((size_t)MROWS * EMB * 2);
    unsigned short* kb  = (unsigned short*)alloc((size_t)MROWS * EMB * 2);
    unsigned short* vtb = (unsigned short*)alloc((size_t)MROWS * EMB * 2);
    unsigned short* ab  = (unsigned short*)alloc((size_t)MROWS * EMB * 2);

    cvt_bf16_kernel<<<1024, 256, 0, stream>>>(x, xb, MROWS * EMB / 8);
    transpose_bf16_kernel<<<1024, 256, 0, stream>>>(Wq, wtq);
    transpose_bf16_kernel<<<1024, 256, 0, stream>>>(Wk, wtk);
    transpose_bf16_kernel<<<1024, 256, 0, stream>>>(Wv, wtv);
    transpose_bf16_kernel<<<1024, 256, 0, stream>>>(Wo, wto);

    // 32 row-blocks * 48 (3 outputs x 16 heads) = 1536 blocks
    qkv_kernel<<<1536, 256, 0, stream>>>(xb, wtq, wtk, wtv, bq, bk, bv, qb, kb, vtb);
    // 64 (b,h) * 128 q-tiles = 8192 waves -> 1024 blocks
    attn_kernel<<<1024, 256, 0, stream>>>(qb, kb, vtb, ab);
    // 32 row-blocks * 16 col-blocks = 512 blocks
    proj_kernel<<<512, 256, 0, stream>>>(ab, wto, bo, out);
}
